// SHDecoder_68771016344037
// MI455X (gfx1250) — matrix operand order repaired and checked by reading, not compiled, not run on hardware
//
#include <hip/hip_runtime.h>
#include <hip/hip_bf16.h>
#include <math.h>

// ---- problem constants (from reference) ----
#define DMODEL 512
#define DINNER 1024
#define NST    16
#define NLAYER 24
#define MROWS  4096          // B*L = 4*1024
#define SEQL   1024

typedef __attribute__((ext_vector_type(16))) __bf16 v16bf;
typedef __attribute__((ext_vector_type(8)))  __bf16 v8bf;
typedef __attribute__((ext_vector_type(4)))  __bf16 v4bf;
typedef __attribute__((ext_vector_type(8)))  float  v8f;

__device__ __forceinline__ float sigm_(float x) { return 1.0f / (1.0f + __expf(-x)); }

// =====================================================================
// bf16 WMMA GEMM:  C[M,N] (+)= A[M,K] * W[N,K]^T  (+ bias[n])
// A,W,C f32 row-major; f32->bf16 cast during LDS staging.
// Block = 128 threads (4 waves). Tile 128x64, K-step 32, double-buffered
// LDS with software pipeline: issue next tile's global loads, run 8 WMMAs
// on current buffer, then convert+store to the alternate buffer.
// Requires K%32==0, M%128==0; N guarded.
// =====================================================================
#define GBM 128
#define GBN 64
#define GBK 32
#define GLD 40   // padded LDS row (bf16 elems): 80B stride -> conflict-free

__global__ __launch_bounds__(128) void k_gemm(
    const float* __restrict__ A, const float* __restrict__ W,
    float* __restrict__ C, const float* __restrict__ bias,
    int M, int N, int K, int lda, int ldw, int ldc, int accumulate)
{
  __shared__ __bf16 As[2][GBM][GLD];
  __shared__ __bf16 Ws[2][GBN][GLD];
  const int tid   = threadIdx.x;
  const int lane  = tid & 31;
  const int wave  = tid >> 5;
  const int khalf = lane >> 4;
  const int l16   = lane & 15;
  const int m0 = blockIdx.y * GBM;
  const int n0 = blockIdx.x * GBN;

  // staging assignments (fixed per thread)
  const int rA  = tid >> 1;                 // iterate +? below: A 1024 f4 / 128 thr = 8
  const int kvA = (tid & 1) << 2;           // base kv, +8 per pair step
  const int rW  = tid >> 3;                 // W 512 f4 / 128 thr = 4 (r stride 16)
  const int kvW = (tid & 7) << 2;

  v8f acc[2][4];
#pragma unroll
  for (int i = 0; i < 2; ++i)
#pragma unroll
    for (int j = 0; j < 4; ++j) acc[i][j] = (v8f)(0.0f);

  float4 fa[8], fw[4];

  auto load_tile = [&](int k0) {
    // A tile: 128 rows x 32 cols = 1024 float4; thread -> rows tid/8 + 16*it;
    // each row 8 float4, this thread takes the float4 at kv = 4*(tid&7) in each:
#pragma unroll
    for (int it = 0; it < 8; ++it) {
      int v  = tid + it * 128;              // 0..1023
      int r  = v >> 3;                      // 0..127
      int kv = (v & 7) << 2;
      fa[it] = *reinterpret_cast<const float4*>(A + (size_t)(m0 + r) * lda + k0 + kv);
    }
#pragma unroll
    for (int it = 0; it < 4; ++it) {
      int v  = tid + it * 128;              // 0..511
      int r  = v >> 3;                      // 0..63
      int kv = (v & 7) << 2;
      float4 f = make_float4(0.f, 0.f, 0.f, 0.f);
      if (n0 + r < N) f = *reinterpret_cast<const float4*>(W + (size_t)(n0 + r) * ldw + k0 + kv);
      fw[it] = f;
    }
  };

  auto store_tile = [&](int buf) {
#pragma unroll
    for (int it = 0; it < 8; ++it) {
      int v  = tid + it * 128;
      int r  = v >> 3;
      int kv = (v & 7) << 2;
      v4bf s; s[0] = (__bf16)fa[it].x; s[1] = (__bf16)fa[it].y;
              s[2] = (__bf16)fa[it].z; s[3] = (__bf16)fa[it].w;
      *reinterpret_cast<v4bf*>(&As[buf][r][kv]) = s;
    }
#pragma unroll
    for (int it = 0; it < 4; ++it) {
      int v  = tid + it * 128;
      int r  = v >> 3;
      int kv = (v & 7) << 2;
      v4bf s; s[0] = (__bf16)fw[it].x; s[1] = (__bf16)fw[it].y;
              s[2] = (__bf16)fw[it].z; s[3] = (__bf16)fw[it].w;
      *reinterpret_cast<v4bf*>(&Ws[buf][r][kv]) = s;
    }
  };

  // prologue: stage first tile
  load_tile(0);
  store_tile(0);
  __syncthreads();

  int cur = 0;
  for (int k0 = 0; k0 < K; k0 += GBK) {
    const int kn = k0 + GBK;
    const bool have_next = (kn < K);
    if (have_next) load_tile(kn);          // issue global loads early (no wait yet)

    // ---- fragments from LDS[cur] ----
    // A fragment layout (16x32 bf16): lanes 0-15 K{0..7,16..23}, 16-31 K{8..15,24..31}
    v16bf afrag[2];
#pragma unroll
    for (int mt = 0; mt < 2; ++mt) {
      const int mr = mt * 64 + wave * 16 + l16;
      v8bf alo = *reinterpret_cast<const v8bf*>(&As[cur][mr][khalf * 8]);
      v8bf ahi = *reinterpret_cast<const v8bf*>(&As[cur][mr][16 + khalf * 8]);
#pragma unroll
      for (int i = 0; i < 8; ++i) { afrag[mt][i] = alo[i]; afrag[mt][i + 8] = ahi[i]; }
    }
    v16bf bfrag[4];
#pragma unroll
    for (int nt = 0; nt < 4; ++nt) {
      // B fragment (32x16): lane holds col n=l16, lanes 0-15 K{0..7,16..23}, 16-31 K{8..15,24..31}
      const int nc = nt * 16 + l16;
      v8bf blo = *reinterpret_cast<const v8bf*>(&Ws[cur][nc][khalf * 8]);
      v8bf bhi = *reinterpret_cast<const v8bf*>(&Ws[cur][nc][16 + khalf * 8]);
#pragma unroll
      for (int i = 0; i < 8; ++i) { bfrag[nt][i] = blo[i]; bfrag[nt][i + 8] = bhi[i]; }
    }

    // ---- 8 back-to-back WMMAs ----
#pragma unroll
    for (int mt = 0; mt < 2; ++mt)
#pragma unroll
      for (int nt = 0; nt < 4; ++nt)
        acc[mt][nt] = __builtin_amdgcn_wmma_f32_16x16x32_bf16(
            false, afrag[mt], false, bfrag[nt], (short)0, acc[mt][nt], false, false);

    if (have_next) store_tile(cur ^ 1);    // waits loads, converts, ds_store
    __syncthreads();
    cur ^= 1;
  }

  // epilogue: C layout lanes0-15 M=r, lanes16-31 M=8+r; N = lane&15
#pragma unroll
  for (int mt = 0; mt < 2; ++mt) {
#pragma unroll
    for (int nt = 0; nt < 4; ++nt) {
      int n = n0 + nt * 16 + l16;
      if (n >= N) continue;
      float bv = bias ? bias[n] : 0.0f;
#pragma unroll
      for (int r = 0; r < 8; ++r) {
        int m = m0 + mt * 64 + wave * 16 + khalf * 8 + r;
        size_t idx = (size_t)m * ldc + n;
        float val = acc[mt][nt][r] + bv;
        C[idx] = accumulate ? (C[idx] + val) : val;
      }
    }
  }
}

// =====================================================================
// Point embedding: x[row, 0:256]=rot, x[row,256:512]=pos
// =====================================================================
__global__ __launch_bounds__(256) void k_embed(
    const float* __restrict__ pose,
    const float* __restrict__ rw, const float* __restrict__ rb,
    const float* __restrict__ pw, const float* __restrict__ pb,
    float* __restrict__ x)
{
  __shared__ float pl[6];
  __shared__ float fr[51];
  __shared__ float fp[51];
  const int row = blockIdx.x, t = threadIdx.x;
  if (t < 6) pl[t] = pose[(size_t)row * 6 + t];
  __syncthreads();
  if (t < 24) {
    int ax = t >> 3, i = t & 7;
    float sc = 3.14159265358979f * (float)(1 << i);
    float pr = pl[ax] * sc, pp = pl[3 + ax] * sc;
    fr[t] = sinf(pr); fr[24 + t] = cosf(pr);
    fp[t] = sinf(pp); fp[24 + t] = cosf(pp);
  }
  if (t < 3) { fr[48 + t] = pl[t]; fp[48 + t] = pl[3 + t]; }
  __syncthreads();
  float ar = rb[t], ap = pb[t];
  for (int k = 0; k < 51; ++k) {
    ar += fr[k] * rw[t * 51 + k];
    ap += fp[k] * pw[t * 51 + k];
  }
  x[(size_t)row * DMODEL + t]       = ar;
  x[(size_t)row * DMODEL + 256 + t] = ap;
}

// =====================================================================
// RMSNorm (512 cols, 256 thr x 2)
// =====================================================================
__global__ __launch_bounds__(256) void k_rmsnorm(
    const float* __restrict__ x, const float* __restrict__ w, float* __restrict__ h)
{
  const int row = blockIdx.x, t = threadIdx.x;
  float v0 = x[(size_t)row * DMODEL + t];
  float v1 = x[(size_t)row * DMODEL + 256 + t];
  __shared__ float red[256];
  red[t] = v0 * v0 + v1 * v1;
  __syncthreads();
  for (int s = 128; s > 0; s >>= 1) { if (t < s) red[t] += red[t + s]; __syncthreads(); }
  float scale = rsqrtf(red[0] * (1.0f / DMODEL) + 1e-5f);
  h[(size_t)row * DMODEL + t]       = v0 * scale * w[t];
  h[(size_t)row * DMODEL + 256 + t] = v1 * scale * w[256 + t];
}

// =====================================================================
// LayerNorm
// =====================================================================
__global__ __launch_bounds__(256) void k_layernorm(
    const float* __restrict__ x, const float* __restrict__ w,
    const float* __restrict__ b, float* __restrict__ h)
{
  const int row = blockIdx.x, t = threadIdx.x;
  float v0 = x[(size_t)row * DMODEL + t];
  float v1 = x[(size_t)row * DMODEL + 256 + t];
  __shared__ float red[256];
  __shared__ float s_mean;
  red[t] = v0 + v1;
  __syncthreads();
  for (int s = 128; s > 0; s >>= 1) { if (t < s) red[t] += red[t + s]; __syncthreads(); }
  if (t == 0) s_mean = red[0] * (1.0f / DMODEL);
  __syncthreads();
  float m = s_mean;
  red[t] = (v0 - m) * (v0 - m) + (v1 - m) * (v1 - m);
  __syncthreads();
  for (int s = 128; s > 0; s >>= 1) { if (t < s) red[t] += red[t + s]; __syncthreads(); }
  float inv = rsqrtf(red[0] * (1.0f / DMODEL) + 1e-5f);
  h[(size_t)row * DMODEL + t]       = (v0 - m) * inv * w[t] + b[t];
  h[(size_t)row * DMODEL + 256 + t] = (v1 - m) * inv * w[256 + t] + b[256 + t];
}

// =====================================================================
// Causal depthwise conv (K=4) + SiLU on xi = xz[:, :1024]
// =====================================================================
__global__ __launch_bounds__(256) void k_conv(
    const float* __restrict__ xz, const float* __restrict__ cw,
    const float* __restrict__ cb, float* __restrict__ xc)
{
  const int m = blockIdx.x, b = m >> 10, l = m & 1023, t = threadIdx.x;
#pragma unroll
  for (int j = 0; j < 4; ++j) {
    int d = t + j * 256;
    float acc = cb[d];
#pragma unroll
    for (int k = 0; k < 4; ++k) {
      int l2 = l + k - 3;
      if (l2 >= 0) acc += cw[d * 4 + k] * xz[((size_t)(b * 1024 + l2)) * (2 * DINNER) + d];
    }
    xc[(size_t)m * DINNER + d] = acc * sigm_(acc);
  }
}

// =====================================================================
// softplus in place (dt buffer, 4096x1024)
// =====================================================================
__global__ __launch_bounds__(256) void k_softplus(float* __restrict__ dt)
{
  const size_t base = (size_t)blockIdx.x * 1024;
  const int t = threadIdx.x;
#pragma unroll
  for (int j = 0; j < 4; ++j) {
    size_t i = base + t + j * 256;
    float x = dt[i];
    dt[i] = (x > 20.0f) ? x : log1pf(expf(x));
  }
}

// =====================================================================
// Selective scan: lane = one (b,d) channel, 16 states in VGPRs.
// grid 16 blocks x 256 threads; B/C staged 64 t-steps at a time into LDS.
// y = scan(dt, B, C, xc) + Dskip*xc
// =====================================================================
__global__ __launch_bounds__(256) void k_scan(
    const float* __restrict__ dt, const float* __restrict__ xc,
    const float* __restrict__ xdbl, const float* __restrict__ Alog,
    const float* __restrict__ Dskip, float* __restrict__ y)
{
  __shared__ float sBC[64][32];
  const int t = threadIdx.x;
  const int b = blockIdx.x >> 2;
  const int d = ((blockIdx.x & 3) << 8) + t;

  float hs[NST], a[NST];
#pragma unroll
  for (int n = 0; n < NST; ++n) {
    hs[n] = 0.0f;
    a[n] = -__expf(Alog[(size_t)d * NST + n]);
  }
  const float Dsk = Dskip[d];

  for (int tc0 = 0; tc0 < SEQL; tc0 += 64) {
    __syncthreads();
#pragma unroll
    for (int it = 0; it < 8; ++it) {
      int v = t + it * 256;
      int tt = v >> 5, c = v & 31;
      sBC[tt][c] = xdbl[((size_t)(b * 1024 + tc0 + tt)) * 64 + 32 + c];
    }
    __syncthreads();
    for (int tl = 0; tl < 64; ++tl) {
      size_t off = (size_t)(b * 1024 + tc0 + tl) * DINNER + d;
      float dtv = dt[off];
      float xv  = xc[off];
      float dx  = dtv * xv;
      float yv  = 0.0f;
#pragma unroll
      for (int n = 0; n < NST; ++n) {
        float e = __expf(dtv * a[n]);
        hs[n] = e * hs[n] + dx * sBC[tl][n];
        yv += hs[n] * sBC[tl][16 + n];
      }
      y[off] = yv + Dsk * xv;
    }
  }
}

// =====================================================================
// gate: y *= silu(z), z = xz[:, 1024:2048]
// =====================================================================
__global__ __launch_bounds__(256) void k_gate(
    float* __restrict__ y, const float* __restrict__ xz)
{
  const int m = blockIdx.x, t = threadIdx.x;
#pragma unroll
  for (int j = 0; j < 4; ++j) {
    int d = t + j * 256;
    float z = xz[(size_t)m * (2 * DINNER) + DINNER + d];
    y[(size_t)m * DINNER + d] *= z * sigm_(z);
  }
}

// =====================================================================
// GEGLU: v = a * gelu_exact(g) from u (biases already fused by GEMM)
// =====================================================================
__global__ __launch_bounds__(256) void k_geglu(
    const float* __restrict__ u, float* __restrict__ v)
{
  const int m = blockIdx.x, t = threadIdx.x;
#pragma unroll
  for (int j = 0; j < 8; ++j) {
    int c = t + j * 256;
    float aa = u[(size_t)m * 4096 + c];
    float g  = u[(size_t)m * 4096 + 2048 + c];
    float ge = 0.5f * g * (1.0f + erff(g * 0.70710678118654752f));
    v[(size_t)m * 2048 + c] = aa * ge;
  }
}

// =====================================================================
static inline void launch_gemm(const float* A, const float* W, float* C,
                               const float* bias, int M, int N, int K,
                               int lda, int ldw, int ldc, int accumulate,
                               hipStream_t s)
{
  dim3 g((N + GBN - 1) / GBN, M / GBM);
  k_gemm<<<g, 128, 0, s>>>(A, W, C, bias, M, N, K, lda, ldw, ldc, accumulate);
}

extern "C" void kernel_launch(void* const* d_in, const int* in_sizes, int n_in,
                              void* d_out, int out_size, void* d_ws, size_t ws_size,
                              hipStream_t stream) {
  (void)in_sizes; (void)n_in; (void)out_size; (void)ws_size;
  const float* pose   = (const float*)d_in[0];
  const float* rot_w  = (const float*)d_in[1];
  const float* rot_b  = (const float*)d_in[2];
  const float* pos_w  = (const float*)d_in[3];
  const float* pos_b  = (const float*)d_in[4];
  const float* norm_w = (const float*)d_in[5];
  const float* Win    = (const float*)d_in[6];
  const float* conv_w = (const float*)d_in[7];
  const float* conv_b = (const float*)d_in[8];
  const float* Wx     = (const float*)d_in[9];
  const float* Wdt    = (const float*)d_in[10];
  const float* bdt    = (const float*)d_in[11];
  const float* A_log  = (const float*)d_in[12];
  const float* Dskip  = (const float*)d_in[13];
  const float* Wout   = (const float*)d_in[14];
  const float* ln_w   = (const float*)d_in[15];
  const float* ln_b   = (const float*)d_in[16];
  const float* ff_w1  = (const float*)d_in[17];
  const float* ff_b1  = (const float*)d_in[18];
  const float* ff_w2  = (const float*)d_in[19];
  const float* ff_b2  = (const float*)d_in[20];
  const float* out_w  = (const float*)d_in[21];
  const float* out_b  = (const float*)d_in[22];
  float* out = (float*)d_out;

  // workspace layout (floats)
  float* ws = (float*)d_ws;
  size_t o = 0;
  float* x    = ws + o; o += (size_t)MROWS * DMODEL;        //  8 MB
  float* h    = ws + o; o += (size_t)MROWS * DMODEL;        //  8 MB
  float* xz   = ws + o; o += (size_t)MROWS * 4096;          // 64 MB (xz uses half; FFN 'u' uses all)
  float* xc   = ws + o; o += (size_t)MROWS * DINNER;        // 16 MB
  float* dtb  = ws + o; o += (size_t)MROWS * DINNER;        // 16 MB
  float* yb   = ws + o; o += (size_t)MROWS * DINNER;        // 16 MB
  float* xdbl = ws + o; o += (size_t)MROWS * 64;            //  1 MB
  float* u    = xz;   // 4096x4096 after layer loop (xz no longer live)
  float* vbuf = xc;   // 4096x2048 spans xc+dtb (both dead in FFN)

  // 1) embeddings -> x
  k_embed<<<MROWS, 256, 0, stream>>>(pose, rot_w, rot_b, pos_w, pos_b, x);

  // 2) 24 Mamba blocks
  for (int l = 0; l < NLAYER; ++l) {
    const float* nw  = norm_w + (size_t)l * DMODEL;
    const float* Wi  = Win    + (size_t)l * 2 * DINNER * DMODEL;
    const float* cw  = conv_w + (size_t)l * DINNER * 4;
    const float* cb  = conv_b + (size_t)l * DINNER;
    const float* Wxl = Wx     + (size_t)l * 64 * DINNER;
    const float* Wdl = Wdt    + (size_t)l * DINNER * 32;
    const float* bdl = bdt    + (size_t)l * DINNER;
    const float* Al  = A_log  + (size_t)l * DINNER * NST;
    const float* Dsl = Dskip  + (size_t)l * DINNER;
    const float* Wo  = Wout   + (size_t)l * DMODEL * DINNER;

    k_rmsnorm<<<MROWS, 256, 0, stream>>>(x, nw, h);
    // xz = h @ Win.T                         (4096 x 2048, K=512)
    launch_gemm(h, Wi, xz, nullptr, MROWS, 2 * DINNER, DMODEL, DMODEL, DMODEL, 2 * DINNER, 0, stream);
    k_conv<<<MROWS, 256, 0, stream>>>(xz, cw, cb, xc);
    // x_dbl = xc @ Wx.T                      (4096 x 64, K=1024)
    launch_gemm(xc, Wxl, xdbl, nullptr, MROWS, 64, DINNER, DINNER, DINNER, 64, 0, stream);
    // dt = x_dbl[:, :32] @ Wdt.T + bdt       (4096 x 1024, K=32)
    launch_gemm(xdbl, Wdl, dtb, bdl, MROWS, DINNER, 32, 64, 32, DINNER, 0, stream);
    k_softplus<<<MROWS, 256, 0, stream>>>(dtb);
    k_scan<<<16, 256, 0, stream>>>(dtb, xc, xdbl, Al, Dsl, yb);
    k_gate<<<MROWS, 256, 0, stream>>>(yb, xz);
    // x += y @ Wout.T                        (4096 x 512, K=1024)
    launch_gemm(yb, Wo, x, nullptr, MROWS, DMODEL, DINNER, DINNER, DINNER, DMODEL, 1, stream);
  }

  // 3) FFN
  k_layernorm<<<MROWS, 256, 0, stream>>>(x, ln_w, ln_b, h);
  // u = h @ ff_w1.T + ff_b1                  (4096 x 4096, K=512)
  launch_gemm(h, ff_w1, u, ff_b1, MROWS, 4096, DMODEL, DMODEL, DMODEL, 4096, 0, stream);
  k_geglu<<<MROWS, 256, 0, stream>>>(u, vbuf);
  // x += v @ ff_w2.T + ff_b2                 (4096 x 512, K=2048)
  launch_gemm(vbuf, ff_w2, x, ff_b2, MROWS, DMODEL, 2048, 2048, 2048, DMODEL, 1, stream);

  // 4) head: out = x @ out_w.T + out_b       (4096 x 16, K=512)
  launch_gemm(x, out_w, out, out_b, MROWS, 16, DMODEL, DMODEL, DMODEL, 16, 0, stream);
}
